// GraphConvolution_82867099009361
// MI455X (gfx1250) — matrix operand order repaired and checked by reading, not compiled, not run on hardware
//
#include <hip/hip_runtime.h>
#include <hip/hip_bf16.h>

#define BATCH 16
#define NN    2048
#define DD    64

typedef __attribute__((ext_vector_type(16))) __bf16 v16bf;
typedef __attribute__((ext_vector_type(8)))  __bf16 v8bf;
typedef __attribute__((ext_vector_type(8)))  float  v8f;
typedef __attribute__((ext_vector_type(4)))  float  v4f;
typedef __attribute__((ext_vector_type(4)))  int    v4i;

__device__ __forceinline__ v8f wmma_bf16(v16bf a, v16bf b, v8f c) {
    // 8 args: (neg_a, A, neg_b, B, c_mod, C, reuse_a, reuse_b)
    return __builtin_amdgcn_wmma_f32_16x16x32_bf16(false, a, false, b, (short)0, c,
                                                   false, false);
}

// A fragment (16x32 bf16): lane = row, chunks k = 8*hi+e and 16+8*hi+e
__device__ __forceinline__ v16bf load_fragA(const __bf16* __restrict__ base,
                                            int row, int ld, int c0, int lane) {
    const int kb = (lane >> 4) * 8;
    const __bf16* p = base + (size_t)row * ld + c0 + kb;
    v8bf lo = *(const v8bf*)(p);
    v8bf hi = *(const v8bf*)(p + 16);
    v16bf f;
#pragma unroll
    for (int i = 0; i < 8; ++i) { f[i] = lo[i]; f[8 + i] = hi[i]; }
    return f;
}

// B fragment (32x16 bf16): lane = column j, chunks k = 8*hi+e and 16+8*hi+e
// B[k][j] = src[j][c0+k], src row-major stride ld (B is a slice of srcT).
__device__ __forceinline__ v16bf load_fragB(const __bf16* __restrict__ base,
                                            int rowj, int ld, int c0, int lane) {
    const int kh = (lane >> 4) * 8;
    const __bf16* p = base + (size_t)rowj * ld + c0 + kh;
    v8bf lo = *(const v8bf*)(p);
    v8bf hi = *(const v8bf*)(p + 16);
    v16bf f;
#pragma unroll
    for (int i = 0; i < 8; ++i) { f[i] = lo[i]; f[8 + i] = hi[i]; }
    return f;
}

// ---------------------------------------------------------------------------
// Kernel 1: inverse norms, bf16 x, and support^T (bf16) via WMMA.
// W^T is staged once per workgroup in LDS as bf16 (conflict-padded), so the
// per-lane B-fragment gather is two aligned ds_load_b128 instead of 32 scalar
// global loads. One wave handles 16 rows of one batch.
// ---------------------------------------------------------------------------
#define WT_LD 72   // 64 + 8 bf16 pad: rows 144 B apart (16B-aligned, bank-skewed)

__global__ __launch_bounds__(256) void gc_prep(const float* __restrict__ x,
                                               const float* __restrict__ W,
                                               __bf16* __restrict__ xh,
                                               __bf16* __restrict__ supT,
                                               float* __restrict__ rinv) {
    __shared__ __bf16 wt[DD * WT_LD];           // Wt[j][k] = W[k][j], 9 KB

    // cooperative transpose W (f32 64x64) -> LDS bf16
    for (int idx = threadIdx.x; idx < DD * DD; idx += 256) {
        const int k = idx >> 6;                 // coalesced read of W row-major
        const int j = idx & 63;
        wt[j * WT_LD + k] = (__bf16)W[idx];
    }
    __syncthreads();

    const int lane = threadIdx.x & 31;
    const int wv   = threadIdx.x >> 5;
    const int gw   = blockIdx.x * 8 + wv;       // 2048 waves total
    const int row0 = gw * 16;
    const int b    = row0 / NN;
    const int n0   = row0 - b * NN;
    const int r    = lane & 15;
    const int hi   = lane >> 4;
    const int kb   = hi * 8;

    const float* xrow = x  + (size_t)(b * NN + n0 + r) * DD;
    __bf16*      hrow = xh + (size_t)(b * NN + n0 + r) * DD;

    const int kcs[4] = {kb, 16 + kb, 32 + kb, 48 + kb};
    v8bf hchunk[4];
    float ss = 0.f;
#pragma unroll
    for (int c = 0; c < 4; ++c) {
        v4f lo = *(const v4f*)(xrow + kcs[c]);
        v4f hv = *(const v4f*)(xrow + kcs[c] + 4);
        v8bf h;
#pragma unroll
        for (int i = 0; i < 4; ++i) {
            ss += lo[i] * lo[i] + hv[i] * hv[i];
            h[i]     = (__bf16)lo[i];
            h[4 + i] = (__bf16)hv[i];
        }
        hchunk[c] = h;
        *(v8bf*)(hrow + kcs[c]) = h;
    }
    ss += __shfl_xor(ss, 16, 32);
    const float rv = rsqrtf(ss);
    if (lane < 16) rinv[b * NN + n0 + lane] = rv;

    v16bf a0, a1;
#pragma unroll
    for (int i = 0; i < 8; ++i) {
        a0[i] = hchunk[0][i]; a0[8 + i] = hchunk[1][i];
        a1[i] = hchunk[2][i]; a1[8 + i] = hchunk[3][i];
    }

#pragma unroll
    for (int jt = 0; jt < 4; ++jt) {
        const int j = jt * 16 + r;
        // B fragment from LDS Wt: lane = column j of W, chunks k = c0 + 8*hi + e and c0 + 16 + 8*hi + e
        const __bf16* wrow = wt + j * WT_LD + 8 * hi;
        v16bf b0, b1;
        {
            v8bf l0 = *(const v8bf*)(wrow);
            v8bf l1 = *(const v8bf*)(wrow + 16);
            v8bf l2 = *(const v8bf*)(wrow + 32);
            v8bf l3 = *(const v8bf*)(wrow + 48);
#pragma unroll
            for (int i = 0; i < 8; ++i) {
                b0[i] = l0[i]; b0[8 + i] = l1[i];
                b1[i] = l2[i]; b1[8 + i] = l3[i];
            }
        }
        v8f acc = {};
        acc = wmma_bf16(a0, b0, acc);
        acc = wmma_bf16(a1, b1, acc);
        v8bf o;
#pragma unroll
        for (int v = 0; v < 8; ++v) o[v] = (__bf16)acc[v];   // n = n0 + 8*hi + v
        *(v8bf*)(supT + ((size_t)b * DD + j) * NN + n0 + 8 * hi) = o;
    }
}

// ---------------------------------------------------------------------------
// Kernel 2: fused  w = (x·xT)·att·mask/(|x||x|) ;  out = w @ support + bias.
// One wave: 16 output rows x 64 output cols; m-loop in chunks of 32.
// ---------------------------------------------------------------------------
__global__ __launch_bounds__(128) void gc_main(const __bf16* __restrict__ xh,
                                               const __bf16* __restrict__ supT,
                                               const float* __restrict__ rinv,
                                               const float* __restrict__ att,
                                               const int*   __restrict__ mask,
                                               const float* __restrict__ bias,
                                               float* __restrict__ out) {
    const int lane = threadIdx.x & 31;
    const int wv   = threadIdx.x >> 5;
    const int b    = blockIdx.x >> 5;
    const int n0   = ((blockIdx.x & 31) * 4 + wv) * 16;
    const int r    = lane & 15;
    const int hi   = lane >> 4;
    const int kb   = 8 * hi;

    __shared__ float sw[4][16][36];               // per-wave 16x32 S tile, padded

    const __bf16* xb = xh   + (size_t)b * NN * DD;
    const __bf16* sb = supT + (size_t)b * DD * NN;
    const float*  rb = rinv + b * NN;

    const v16bf a0 = load_fragA(xb, n0 + r, DD, 0,  lane);
    const v16bf a1 = load_fragA(xb, n0 + r, DD, 32, lane);
    const float rn = rb[n0 + r];
    const float*  attrow  = att  + (size_t)(n0 + r) * NN;
    const int*    maskrow = mask + (size_t)(n0 + r) * NN;

    v8f acc0 = {}, acc1 = {}, acc2 = {}, acc3 = {};

    for (int m0 = 0; m0 < NN; m0 += 32) {
        __builtin_prefetch(attrow + m0 + 32, 0, 1);     // global_prefetch_b8

        // GEMM1: S = Xn(16x64) * Xm^T for two 16-wide m subtiles
#pragma unroll
        for (int t = 0; t < 2; ++t) {
            v16bf bB0 = load_fragB(xb, m0 + t * 16 + r, DD, 0,  lane);
            v16bf bB1 = load_fragB(xb, m0 + t * 16 + r, DD, 32, lane);
            v8f s = {};
            s = wmma_bf16(a0, bB0, s);
            s = wmma_bf16(a1, bB1, s);
#pragma unroll
            for (int v = 0; v < 8; ++v)                 // D layout: M=v+8*hi, N=r
                sw[wv][v + 8 * hi][t * 16 + r] = s[v];
        }
        asm volatile("s_wait_dscnt 0" ::: "memory");    // in-wave LDS transpose fence

        // Relayout D -> A (16x32) with fused elementwise weighting
        v16bf aw;
#pragma unroll
        for (int c = 0; c < 2; ++c) {
            const int mb = c * 16 + kb;
            v4f s0  = *(const v4f*)&sw[wv][r][mb];
            v4f s1  = *(const v4f*)&sw[wv][r][mb + 4];
            v4f at0 = *(const v4f*)(attrow + m0 + mb);
            v4f at1 = *(const v4f*)(attrow + m0 + mb + 4);
            v4i mk0 = *(const v4i*)(maskrow + m0 + mb);
            v4i mk1 = *(const v4i*)(maskrow + m0 + mb + 4);
            v4f rm0 = *(const v4f*)(rb + m0 + mb);
            v4f rm1 = *(const v4f*)(rb + m0 + mb + 4);
#pragma unroll
            for (int i = 0; i < 4; ++i) {
                float w0 = s0[i] * at0[i] * (float)mk0[i] * rn * rm0[i];
                float w1 = s1[i] * at1[i] * (float)mk1[i] * rn * rm1[i];
                aw[c * 8 + i]     = (__bf16)w0;
                aw[c * 8 + 4 + i] = (__bf16)w1;
            }
        }
        asm volatile("" ::: "memory");   // DS ops in-order per wave; just block reordering

        // GEMM2: out += w(16x32) @ support[m0..m0+31][:]
        v16bf b0 = load_fragB(sb,  0 + r, NN, m0, lane);
        v16bf b1 = load_fragB(sb, 16 + r, NN, m0, lane);
        v16bf b2 = load_fragB(sb, 32 + r, NN, m0, lane);
        v16bf b3 = load_fragB(sb, 48 + r, NN, m0, lane);
        acc0 = wmma_bf16(aw, b0, acc0);
        acc1 = wmma_bf16(aw, b1, acc1);
        acc2 = wmma_bf16(aw, b2, acc2);
        acc3 = wmma_bf16(aw, b3, acc3);
    }

    const v8f accs[4] = {acc0, acc1, acc2, acc3};
#pragma unroll
    for (int jt = 0; jt < 4; ++jt) {
        const int j = jt * 16 + r;
        const float bv = bias[j];
#pragma unroll
        for (int v = 0; v < 8; ++v) {
            const int n = n0 + v + 8 * hi;
            out[((size_t)b * NN + n) * DD + j] = accs[jt][v] + bv;
        }
    }
}

extern "C" void kernel_launch(void* const* d_in, const int* in_sizes, int n_in,
                              void* d_out, int out_size, void* d_ws, size_t ws_size,
                              hipStream_t stream) {
    const float* x    = (const float*)d_in[0];
    const float* W    = (const float*)d_in[1];
    const float* att  = (const float*)d_in[2];
    const float* bias = (const float*)d_in[3];
    const int*   mask = (const int*)d_in[4];
    float* out = (float*)d_out;

    char* ws = (char*)d_ws;
    __bf16* xh   = (__bf16*)(ws);                                   // 4 MB
    __bf16* supT = (__bf16*)(ws + (size_t)BATCH * NN * DD * 2);     // 4 MB
    float*  rinv = (float*) (ws + (size_t)BATCH * NN * DD * 4);     // 128 KB

    gc_prep<<<256, 256, 0, stream>>>(x, W, xh, supT, rinv);
    gc_main<<<512, 128, 0, stream>>>(xh, supT, rinv, att, mask, bias, out);
}
